// VectorQuantizer_6897717477701
// MI455X (gfx1250) — matrix operand order repaired and checked by reading, not compiled, not run on hardware
//
#include <hip/hip_runtime.h>

typedef __attribute__((ext_vector_type(16))) __bf16 v16bf;
typedef __attribute__((ext_vector_type(8)))  float  v8f;
typedef __attribute__((ext_vector_type(4)))  int    v4i;

#define D_DIM 256
#define CH2 64            // codebook rows per LDS chunk (double-buffered)
#define LDS_STRIDE 264    // bf16 elems per LDS row (256 + 8 pad; 528B = 33*16B, 16B-aligned)

#if defined(__AMDGCN__) && __has_builtin(__builtin_amdgcn_global_load_async_to_lds_b128)
#define VQ_ASYNC 1
#else
#define VQ_ASYNC 0
#endif

#if VQ_ASYNC
typedef __attribute__((address_space(1))) v4i* vq_gptr;   // global int4*
typedef __attribute__((address_space(3))) v4i* vq_lptr;   // LDS int4*
#endif

__device__ __forceinline__ unsigned short f32_to_bf16(float f) {
    unsigned int x = __float_as_uint(f);
    unsigned int r = x + 0x7FFFu + ((x >> 16) & 1u);   // round-to-nearest-even
    return (unsigned short)(r >> 16);
}
__device__ __forceinline__ float bf16_to_f32(unsigned short h) {
    return __uint_as_float(((unsigned int)h) << 16);
}

struct B32 { uint4 a, b; };            // 32 bytes -> one v16bf fragment
struct P16 { unsigned int w[8]; };     // 8 packed bf16 pairs -> one v16bf fragment

__device__ __forceinline__ void vq_wait_async() {
#if VQ_ASYNC
#if __has_builtin(__builtin_amdgcn_s_wait_asynccnt)
    __builtin_amdgcn_s_wait_asynccnt(0);
#else
    asm volatile("s_wait_asynccnt 0x0" ::: "memory");
#endif
#endif
}

// Stage CH2 codebook rows (hi+lo bf16) into one LDS buffer.
__device__ __forceinline__ void vq_prefetch_chunk(const unsigned short* __restrict__ cb_hi,
                                                  const unsigned short* __restrict__ cb_lo,
                                                  unsigned short* __restrict__ buf,  // hi; lo at +CH2*LDS_STRIDE
                                                  int chunk, int tid) {
    const int units = CH2 * (D_DIM / 8);   // 16-byte units per array = 2048
#if VQ_ASYNC
    for (int u = tid; u < units; u += 256) {
        int r = u >> 5, c16 = u & 31;
        const unsigned short* gh = cb_hi + (size_t)(chunk + r) * D_DIM + c16 * 8;
        const unsigned short* gl = cb_lo + (size_t)(chunk + r) * D_DIM + c16 * 8;
        unsigned short* lh = buf + r * LDS_STRIDE + c16 * 8;
        unsigned short* ll = lh + CH2 * LDS_STRIDE;
        __builtin_amdgcn_global_load_async_to_lds_b128((vq_gptr)(unsigned short*)gh, (vq_lptr)lh, 0, 0);
        __builtin_amdgcn_global_load_async_to_lds_b128((vq_gptr)(unsigned short*)gl, (vq_lptr)ll, 0, 0);
    }
#else
    for (int u = tid; u < units; u += 256) {
        int r = u >> 5, c16 = u & 31;
        uint4 vh = *(const uint4*)(cb_hi + (size_t)(chunk + r) * D_DIM + c16 * 8);
        uint4 vl = *(const uint4*)(cb_lo + (size_t)(chunk + r) * D_DIM + c16 * 8);
        *(uint4*)(buf + r * LDS_STRIDE + c16 * 8) = vh;
        *(uint4*)(buf + CH2 * LDS_STRIDE + r * LDS_STRIDE + c16 * 8) = vl;
    }
#endif
}

// ---------------- prep 1: codebook fp32 -> bf16 hi/lo (split precision) ----------------
__global__ void vq_prep_convert(const float* __restrict__ cb,
                                unsigned short* __restrict__ hi,
                                unsigned short* __restrict__ lo,
                                int total) {
    int i = (blockIdx.x * blockDim.x + threadIdx.x) * 4;
    if (i >= total) return;
    float4 v = *(const float4*)(cb + i);
    float f[4] = {v.x, v.y, v.z, v.w};
    ushort4 h, l;
    unsigned short hh[4], ll[4];
#pragma unroll
    for (int j = 0; j < 4; ++j) {
        hh[j] = f32_to_bf16(f[j]);
        ll[j] = f32_to_bf16(f[j] - bf16_to_f32(hh[j]));
    }
    h.x = hh[0]; h.y = hh[1]; h.z = hh[2]; h.w = hh[3];
    l.x = ll[0]; l.y = ll[1]; l.z = ll[2]; l.w = ll[3];
    *(ushort4*)(hi + i) = h;
    *(ushort4*)(lo + i) = l;
}

// ---------------- prep 2: c2[k] = ||codebook[k]||^2 (deterministic, no atomics) --------
__global__ void vq_prep_c2(const float* __restrict__ cb, float* __restrict__ c2, int K) {
    int k = blockIdx.x * blockDim.x + threadIdx.x;
    if (k >= K) return;
    const float* r = cb + (size_t)k * D_DIM;
    float s = 0.f;
#pragma unroll 4
    for (int j = 0; j < D_DIM; j += 4) {
        float4 v = *(const float4*)(r + j);
        s += v.x * v.x + v.y * v.y + v.z * v.z + v.w * v.w;
    }
    c2[k] = s;
}

// ---------------- main: WMMA distance GEMM + argmin + gather ---------------------------
__global__ void __launch_bounds__(256)
vq_main(const float* __restrict__ x,
        const float* __restrict__ cb,
        const unsigned short* __restrict__ cb_hi,
        const unsigned short* __restrict__ cb_lo,
        const float* __restrict__ c2,
        float* __restrict__ out,
        int K) {
    extern __shared__ unsigned char smem[];
    float*          s_c2  = (float*)smem;                               // K floats
    unsigned short* s_buf = (unsigned short*)(smem + (size_t)K * sizeof(float));
    const int       bufElems = 2 * CH2 * LDS_STRIDE;                    // hi+lo per buffer

    const int tid  = threadIdx.x;
    const int wave = tid >> 5;
    const int lane = tid & 31;
    const int nl   = lane & 15;     // M index for A, N index for B/C
    const int sel  = lane >> 4;     // K sub-block selector

    // stage c2 into LDS once
    for (int i = tid; i < K; i += 256) s_c2[i] = c2[i];

    // ---- load + convert this wave's 16 input rows into A fragments (registers) ----
    // ISA 16-bit A 16x32 layout: lane m = nl; elems 0..7  -> K = 32s + 8*sel + {0..7}
    //                                        elems 8..15 -> K = 32s + 8*sel + 16 + {0..7}
    const int rowBase = blockIdx.x * 128 + wave * 16;
    const float* xrow = x + (size_t)(rowBase + nl) * D_DIM;
    v16bf Ahi[8], Alo[8];
#pragma unroll
    for (int s = 0; s < 8; ++s) {
        const float* p = xrow + 32 * s + 8 * sel;
        P16 ph, pl;
#pragma unroll
        for (int c = 0; c < 2; ++c) {
            float4 v0 = *(const float4*)(p + 16 * c);
            float4 v1 = *(const float4*)(p + 16 * c + 4);
            float f[8] = {v0.x, v0.y, v0.z, v0.w, v1.x, v1.y, v1.z, v1.w};
#pragma unroll
            for (int j = 0; j < 4; ++j) {
                unsigned short h0 = f32_to_bf16(f[2 * j]);
                unsigned short h1 = f32_to_bf16(f[2 * j + 1]);
                unsigned short l0 = f32_to_bf16(f[2 * j]     - bf16_to_f32(h0));
                unsigned short l1 = f32_to_bf16(f[2 * j + 1] - bf16_to_f32(h1));
                ph.w[4 * c + j] = (unsigned int)h0 | ((unsigned int)h1 << 16);
                pl.w[4 * c + j] = (unsigned int)l0 | ((unsigned int)l1 << 16);
            }
        }
        Ahi[s] = __builtin_bit_cast(v16bf, ph);
        Alo[s] = __builtin_bit_cast(v16bf, pl);
    }

    float best[8];
    int   bidx[8];
#pragma unroll
    for (int i = 0; i < 8; ++i) { best[i] = 3.4e38f; bidx[i] = 0; }

    // ---- double-buffered codebook stream: async global->LDS overlapped with compute ----
    const int nchunks = K / CH2;                 // 16
    vq_prefetch_chunk(cb_hi, cb_lo, s_buf, 0, tid);

    for (int c = 0; c < nchunks; ++c) {
        const int b = c & 1;
        vq_wait_async();                         // my async loads into buf[b] landed
        __syncthreads();                         // everyone's loads landed / reads of buf[1-b] done
        if (c + 1 < nchunks)
            vq_prefetch_chunk(cb_hi, cb_lo, s_buf + (size_t)(1 - b) * bufElems,
                              (c + 1) * CH2, tid);

        const unsigned short* bufh = s_buf + (size_t)b * bufElems;
        for (int t = 0; t < CH2 / 16; ++t) {
            const int nloc = t * 16 + nl;
            const unsigned short* bh = bufh + nloc * LDS_STRIDE;
            const unsigned short* bl = bh + CH2 * LDS_STRIDE;
            v8f acc = {};
#pragma unroll
            for (int s = 0; s < 8; ++s) {
                // ISA 16-bit B 32x16 layout: lane n = nl; elems 0..7 -> K = 32s + 8*sel + e, elems 8..15 -> K = 32s + 8*sel + 16 + (e-8)
                const int off = 32 * s + 8 * sel;
                B32 qh, ql;
                qh.a = *(const uint4*)(bh + off);
                qh.b = *(const uint4*)(bh + off + 16);
                ql.a = *(const uint4*)(bl + off);
                ql.b = *(const uint4*)(bl + off + 16);
                v16bf Bhi = __builtin_bit_cast(v16bf, qh);
                v16bf Blo = __builtin_bit_cast(v16bf, ql);
                acc = __builtin_amdgcn_wmma_f32_16x16x32_bf16(false, Ahi[s], false, Bhi, (short)0, acc, false, false);
                acc = __builtin_amdgcn_wmma_f32_16x16x32_bf16(false, Ahi[s], false, Blo, (short)0, acc, false, false);
                acc = __builtin_amdgcn_wmma_f32_16x16x32_bf16(false, Alo[s], false, Bhi, (short)0, acc, false, false);
            }
            const int   n   = c * CH2 + nloc;
            const float c2n = s_c2[n];
            // C layout: VGPR i holds row (i + 8*sel), column nl; score = c2 - 2*dot
#pragma unroll
            for (int i = 0; i < 8; ++i) {
                float sc = fmaf(-2.0f, acc[i], c2n);
                if (sc < best[i]) { best[i] = sc; bidx[i] = n; }
            }
        }
        __syncthreads();                         // done reading buf[b] before it is refilled
    }

    // ---- cross-lane argmin reduction within each 16-lane half (N is lane-striped) ----
#pragma unroll
    for (int m = 1; m < 16; m <<= 1) {
#pragma unroll
        for (int i = 0; i < 8; ++i) {
            float ob = __shfl_xor(best[i], m, 32);
            int   oi = __shfl_xor(bidx[i], m, 32);
            if (ob < best[i] || (ob == best[i] && oi < bidx[i])) { best[i] = ob; bidx[i] = oi; }
        }
    }

    // ---- gather winning fp32 codebook rows -> output (coalesced float4) ----
#pragma unroll
    for (int r = 0; r < 16; ++r) {
        int k = __shfl(bidx[r & 7], (r < 8) ? 0 : 16, 32);
        const float* src = cb + (size_t)k * D_DIM;
        float*       dst = out + (size_t)(rowBase + r) * D_DIM;
        for (int cc = lane * 4; cc < D_DIM; cc += 128) {
            *(float4*)(dst + cc) = *(const float4*)(src + cc);
        }
    }
}

// ---------------------------------------------------------------------------------------
extern "C" void kernel_launch(void* const* d_in, const int* in_sizes, int n_in,
                              void* d_out, int out_size, void* d_ws, size_t ws_size,
                              hipStream_t stream) {
    const float* x  = (const float*)d_in[0];   // [B*T, 256] fp32
    const float* cb = (const float*)d_in[1];   // [K, 256]   fp32
    float* out = (float*)d_out;                // [B*T, 256] fp32

    const int n0   = in_sizes[0];
    const int n1   = in_sizes[1];
    const int rows = n0 / D_DIM;               // 65536
    const int K    = n1 / D_DIM;               // 1024

    // workspace: [c2: K floats][cb_hi: K*D bf16][cb_lo: K*D bf16]  (~1.03 MB)
    float*          c2 = (float*)d_ws;
    unsigned short* hi = (unsigned short*)((char*)d_ws + (size_t)K * sizeof(float));
    unsigned short* lo = hi + (size_t)K * D_DIM;

    vq_prep_convert<<<(n1 / 4 + 255) / 256, 256, 0, stream>>>(cb, hi, lo, n1);
    vq_prep_c2<<<(K + 255) / 256, 256, 0, stream>>>(cb, c2, K);

    const size_t shmem = (size_t)K * sizeof(float)
                       + 2u * 2u * CH2 * LDS_STRIDE * sizeof(unsigned short);  // ~136 KB
    (void)hipFuncSetAttribute((const void*)vq_main, hipFuncAttributeMaxDynamicSharedMemorySize, (int)shmem);

    vq_main<<<rows / 128, 256, shmem, stream>>>(x, cb, hi, lo, c2, out, K);
}
